// VisionTransformer_8297876815915
// MI455X (gfx1250) — compile-verified, matrix operand order checked by reading, not run on hardware
//
#include <hip/hip_runtime.h>
#include <math.h>

// ---------------- problem constants ----------------
#define TOK      197
#define NSEQP    224          // 197 padded to 7*32
#define DMODEL   768
#define NHEAD    12
#define HDIM     32
#define MLPD     3072
#define NLAYER   12
#define BATCH    32
#define NCLS     1000
#define ROWS_TOTAL (BATCH * TOK)   // 6304
#define PATCH_ROWS (BATCH * 196)   // 6272

// ---------------- WMMA plumbing ----------------
typedef __attribute__((ext_vector_type(16))) __bf16        v16bf;
typedef __attribute__((ext_vector_type(8)))  float         v8f;
typedef __attribute__((ext_vector_type(4)))  unsigned int  u32x4;
typedef __attribute__((ext_vector_type(8)))  int           i32x8;
typedef __attribute__((ext_vector_type(4)))  int           i32x4;

#if defined(__HIP_DEVICE_COMPILE__) && __has_builtin(__builtin_amdgcn_tensor_load_to_lds) && \
    __has_builtin(__builtin_amdgcn_s_wait_tensorcnt)
#define USE_TDM 1
#else
#define USE_TDM 0
#endif

union Frag {
  u32x4          q[2];
  v16bf          v;
  unsigned short u[16];
};

__device__ __forceinline__ unsigned short f32_to_bf16(float f) {
  unsigned int u = __float_as_uint(f);
  u += 0x7FFFu + ((u >> 16) & 1u);   // round-to-nearest-even
  return (unsigned short)(u >> 16);
}
__device__ __forceinline__ float bf16_to_f32(unsigned short h) {
  return __uint_as_float(((unsigned int)h) << 16);
}
__device__ __forceinline__ float gelu_tanh(float x) {
  return 0.5f * x * (1.0f + tanhf(0.7978845608028654f * (x + 0.044715f * x * x * x)));
}
__device__ __forceinline__ v8f wmma_bf16(const Frag& a, const Frag& b, v8f c) {
  // D = A(16x32 bf16) * B(32x16 bf16) + C(16x16 f32)
  return __builtin_amdgcn_wmma_f32_16x16x32_bf16(false, a.v, false, b.v,
                                                 (short)0, c, false, false);
}

// ---------------- elementwise helpers ----------------
__global__ void convert_bf16_kernel(const float* __restrict__ in,
                                    unsigned short* __restrict__ out, long long n) {
  long long i = (long long)blockIdx.x * blockDim.x + threadIdx.x;
  long long stride = (long long)gridDim.x * blockDim.x;
  for (; i < n; i += stride) out[i] = f32_to_bf16(in[i]);
}

// in: [R][C] f32  ->  out: [C][R] bf16
__global__ void transpose_bf16_kernel(const float* __restrict__ in,
                                      unsigned short* __restrict__ out, int R, int C) {
  int idx = blockIdx.x * blockDim.x + threadIdx.x;
  if (idx < R * C) {
    int r = idx / C, c = idx % C;
    out[(size_t)c * R + r] = f32_to_bf16(in[idx]);
  }
}

// im2col for 16x16 stride-16 conv on NCHW [B,3,224,224] -> [B*196][768] bf16
// row = b*196 + py*14 + px ; col = c*256 + ph*16 + pw
__global__ void im2col_kernel(const float* __restrict__ x,
                              unsigned short* __restrict__ out) {
  int row = blockIdx.x;
  int b = row / 196, t = row % 196;
  int py = t / 14, px = t % 14;
  for (int e = threadIdx.x; e < DMODEL; e += 256) {
    int c = e >> 8, r = e & 255;
    int ph = r >> 4, pw = r & 15;
    float v = x[((size_t)(b * 3 + c) * 224 + py * 16 + ph) * 224 + px * 16 + pw];
    out[(size_t)row * DMODEL + e] = f32_to_bf16(v);
  }
}

// h[b][n][d] = pos[n][d] + (n==0 ? cls[d] : pe[(b*196 + n-1)][d])
__global__ void assemble_h_kernel(const float* __restrict__ pe,
                                  const float* __restrict__ cls,
                                  const float* __restrict__ pos,
                                  float* __restrict__ h) {
  long long idx = (long long)blockIdx.x * blockDim.x + threadIdx.x;
  long long total = (long long)ROWS_TOTAL * DMODEL;
  if (idx >= total) return;
  int d = (int)(idx % DMODEL);
  long long bn = idx / DMODEL;
  int n = (int)(bn % TOK);
  int b = (int)(bn / TOK);
  float v = pos[(size_t)n * DMODEL + d];
  v += (n == 0) ? cls[d] : pe[((size_t)b * 196 + n - 1) * DMODEL + d];
  h[idx] = v;
}

__global__ void gather_cls_kernel(const float* __restrict__ h, float* __restrict__ out) {
  int idx = blockIdx.x * blockDim.x + threadIdx.x;
  if (idx < BATCH * DMODEL) {
    int b = idx / DMODEL, d = idx % DMODEL;
    out[idx] = h[(size_t)b * TOK * DMODEL + d];
  }
}

// lambda[i] = exp(lq1.lk1) - exp(lq2.lk2) + lam_init(i)
__global__ void lambda_kernel(const float* __restrict__ lq1, const float* __restrict__ lk1,
                              const float* __restrict__ lq2, const float* __restrict__ lk2,
                              float* __restrict__ lam) {
  int layer = blockIdx.x, t = threadIdx.x;   // 32 threads == HDIM
  float a = lq1[layer * HDIM + t] * lk1[layer * HDIM + t];
  float c = lq2[layer * HDIM + t] * lk2[layer * HDIM + t];
#pragma unroll
  for (int off = 16; off; off >>= 1) {
    a += __shfl_xor(a, off, 32);
    c += __shfl_xor(c, off, 32);
  }
  if (t == 0) {
    float li = 0.8f - 0.6f * __expf(-0.3f * (float)layer);
    lam[layer] = __expf(a) - __expf(c) + li;
  }
}

// ---------------- layernorm: f32 row -> bf16 row ----------------
__global__ __launch_bounds__(256) void layernorm_kernel(const float* __restrict__ x,
                                                        const float* __restrict__ s,
                                                        const float* __restrict__ bsh,
                                                        unsigned short* __restrict__ out) {
  const int Ddim = DMODEL;
  int row = blockIdx.x;
  const float* xr = x + (size_t)row * Ddim;
  float sum = 0.0f, sq = 0.0f;
  for (int i = threadIdx.x; i < Ddim; i += 256) {
    float v = xr[i];
    sum += v; sq += v * v;
  }
#pragma unroll
  for (int off = 16; off; off >>= 1) {
    sum += __shfl_xor(sum, off, 32);
    sq  += __shfl_xor(sq, off, 32);
  }
  __shared__ float s1[8], s2[8];
  int wave = threadIdx.x >> 5, lane = threadIdx.x & 31;
  if (lane == 0) { s1[wave] = sum; s2[wave] = sq; }
  __syncthreads();
  if (wave == 0) {
    float a = (lane < 8) ? s1[lane] : 0.0f;
    float c = (lane < 8) ? s2[lane] : 0.0f;
#pragma unroll
    for (int off = 4; off; off >>= 1) {
      a += __shfl_xor(a, off, 8);
      c += __shfl_xor(c, off, 8);
    }
    if (lane == 0) {
      float mu = a / Ddim;
      s1[0] = mu;
      s2[0] = c / Ddim - mu * mu;
    }
  }
  __syncthreads();
  float mu = s1[0];
  float inv = rsqrtf(s2[0] + 1e-5f);
  for (int i = threadIdx.x; i < Ddim; i += 256)
    out[(size_t)row * Ddim + i] = f32_to_bf16((xr[i] - mu) * inv * s[i] + bsh[i]);
}

// ---------------- generic bf16 WMMA GEMM ----------------
// C[M,N] = act(A[M,K] @ B[K,N] + bias) + residual ; outputs f32 and/or bf16
// A tile is staged by the Tensor Data Mover (TDM) when available:
//  - 2D D#: 128x32 bf16 tile, pad_enable adds 16B per 64B row -> LDS stride 40 halves
//  - one wave issues TENSOR_LOAD_TO_LDS, waits on TENSORcnt, barrier publishes
#define GT_BM 128
#define GT_BN 128
#define GT_BK 32
#define LDS_STR 40   // 32 halves + 8 pad (80B rows, 16B aligned)

__global__ __launch_bounds__(256) void gemm_bf16_kernel(
    const unsigned short* __restrict__ A,
    const unsigned short* __restrict__ Bw,
    const float* __restrict__ bias,
    const float* __restrict__ residual,
    float* __restrict__ Cf,
    unsigned short* __restrict__ Cbf,
    int Mdim, int Ndim, int Kdim, int act)
{
  __shared__ alignas(16) unsigned short sA[GT_BM * LDS_STR];
  __shared__ alignas(16) unsigned short sB[GT_BN * LDS_STR];   // transposed: sB[n][k]

  const int tid = threadIdx.x;
  const int wave = tid >> 5, lane = tid & 31;
  const int wr = wave >> 2;              // 0..1  -> 64-row half
  const int wc = wave & 3;               // 0..3  -> 32-col quarter
  const int m_lane = lane & 15;
  const int kb = (lane >> 4) << 3;       // 0 or 8 (bf16 A/B lane layout)
  const int bm = blockIdx.y * GT_BM;
  const int bn = blockIdx.x * GT_BN;

  const int a_row = tid >> 1;
  const int a_col = (tid & 1) << 4;
  const int b_k   = tid >> 3;
  const int b_n   = (tid & 7) << 4;

  v8f acc[4][2];
#pragma unroll
  for (int i = 0; i < 4; ++i)
#pragma unroll
    for (int j = 0; j < 2; ++j) acc[i][j] = (v8f)0.0f;

  for (int k0 = 0; k0 < Kdim; k0 += GT_BK) {
#if USE_TDM
    // ---- stage A tile [128][32] via Tensor Data Mover ----
    if (tid == 0) {
      unsigned lds_off = (unsigned)(size_t)(void*)sA;
      unsigned long long gaddr =
          (unsigned long long)(size_t)(A + (size_t)bm * Kdim + k0);
      // D# group 0: count=1 | lds_addr | global_addr(57b) | type=2
      u32x4 g0;
      g0.x = 1u;
      g0.y = lds_off;
      g0.z = (unsigned)(gaddr & 0xFFFFFFFFull);
      g0.w = (unsigned)((gaddr >> 32) & 0x01FFFFFFull) | (2u << 30);
      // D# group 1: data_size=2B, pad 16B per 64B row, dims/strides, tile 32x128
      unsigned td0 = (unsigned)(Kdim - k0);   // remaining elems in dim0
      unsigned td1 = (unsigned)(Mdim - bm);   // remaining rows
      unsigned long long st0 = (unsigned long long)(unsigned)Kdim;
      i32x8 g1;
      g1[0] = (int)((1u << 16) | (1u << 20) | (3u << 22) | (3u << 25));
      g1[1] = (int)((td0 & 0xFFFFu) << 16);
      g1[2] = (int)((td0 >> 16) | ((td1 & 0xFFFFu) << 16));
      g1[3] = (int)((td1 >> 16) | ((unsigned)GT_BK << 16));   // tile_dim0 = 32
      g1[4] = (int)(unsigned)GT_BM;                           // tile_dim1 = 128
      g1[5] = (int)(unsigned)(st0 & 0xFFFFFFFFull);           // dim0 stride lo
      g1[6] = (int)(unsigned)((st0 >> 32) & 0xFFFFu);         // dim0 stride hi
      g1[7] = 0;
      i32x4 g2 = (i32x4)0;
      i32x4 g3 = (i32x4)0;
      i32x8 g4 = (i32x8)0;   // extended group (6-arg clang-23 form), unused: zeros
      __builtin_amdgcn_tensor_load_to_lds(g0, g1, g2, g3, g4, 0);
      __builtin_amdgcn_s_wait_tensorcnt(0);
    }
#else
    { // stage A tile [128][32] (manual fallback)
      u32x4 d0 = (u32x4)0u, d1 = (u32x4)0u;
      int gr = bm + a_row;
      if (gr < Mdim) {
        const u32x4* src = (const u32x4*)(A + (size_t)gr * Kdim + k0 + a_col);
        d0 = src[0];
        d1 = src[1];
        if (k0 + GT_BK < Kdim)
          __builtin_prefetch(A + (size_t)gr * Kdim + k0 + GT_BK + a_col, 0, 1);
      }
      u32x4* dst = (u32x4*)(sA + a_row * LDS_STR + a_col);
      dst[0] = d0; dst[1] = d1;
    }
#endif
    { // stage B tile [32][128], transposed into sB[n][k]
      Frag t;
      int gk = k0 + b_k;
      int gn0 = bn + b_n;
      if (gn0 + 16 <= Ndim) {
        const u32x4* src = (const u32x4*)(Bw + (size_t)gk * Ndim + gn0);
        t.q[0] = src[0]; t.q[1] = src[1];
      } else {
#pragma unroll
        for (int j = 0; j < 16; ++j) {
          int gn = gn0 + j;
          t.u[j] = (gn < Ndim) ? Bw[(size_t)gk * Ndim + gn] : (unsigned short)0;
        }
      }
#pragma unroll
      for (int j = 0; j < 16; ++j)
        sB[(b_n + j) * LDS_STR + b_k] = t.u[j];
    }
    __syncthreads();

    Frag af[4], bfr[2];
#pragma unroll
    for (int ms = 0; ms < 4; ++ms) {
      const unsigned short* p = sA + (wr * 64 + ms * 16 + m_lane) * LDS_STR + kb;
      af[ms].q[0] = *(const u32x4*)p;
      af[ms].q[1] = *(const u32x4*)(p + 16);
    }
#pragma unroll
    for (int ns = 0; ns < 2; ++ns) {
      const unsigned short* p = sB + (wc * 32 + ns * 16 + m_lane) * LDS_STR + kb;
      bfr[ns].q[0] = *(const u32x4*)p;
      bfr[ns].q[1] = *(const u32x4*)(p + 16);
    }
#pragma unroll
    for (int ms = 0; ms < 4; ++ms)
#pragma unroll
      for (int ns = 0; ns < 2; ++ns)
        acc[ms][ns] = wmma_bf16(af[ms], bfr[ns], acc[ms][ns]);
    __syncthreads();
  }

  // epilogue: C layout -> lane col = lane&15, row = r + 8*(lane>>4)
  const int hi = lane >> 4;
#pragma unroll
  for (int ms = 0; ms < 4; ++ms) {
#pragma unroll
    for (int ns = 0; ns < 2; ++ns) {
      int gc = bn + wc * 32 + ns * 16 + m_lane;
      if (gc >= Ndim) continue;
      float bv = bias ? bias[gc] : 0.0f;
#pragma unroll
      for (int r = 0; r < 8; ++r) {
        int gr = bm + wr * 64 + ms * 16 + hi * 8 + r;
        if (gr >= Mdim) continue;
        float v = acc[ms][ns][r] + bv;
        if (act == 1) v = gelu_tanh(v);
        size_t idx = (size_t)gr * Ndim + gc;
        if (residual) v += residual[idx];
        if (Cf)  Cf[idx]  = v;
        if (Cbf) Cbf[idx] = f32_to_bf16(v);
      }
    }
  }
}

// ---------------- differential attention ----------------
// grid: (ceil(197/32)=7, NHEAD, BATCH); block 256 (8 waves)
__global__ __launch_bounds__(256) void diff_attn_kernel(
    const unsigned short* __restrict__ Q,   // [B*197][768] bf16
    const unsigned short* __restrict__ K,
    const unsigned short* __restrict__ V,
    const float* __restrict__ lam_buf,      // [L]
    const float* __restrict__ subln,        // [64] for this layer
    float lam_init, int layer,
    unsigned short* __restrict__ Out)       // [B*197][768] bf16
{
  __shared__ alignas(16) float          sS[32 * NSEQP];          // raw scores (f32)
  __shared__ alignas(16) unsigned short sP[32 * (NSEQP + 8)];    // probs / diff-attn (bf16)
  __shared__ alignas(16) float          sO[32 * 64];             // output tile

  const int b = blockIdx.z, h = blockIdx.y, rc = blockIdx.x;
  const int row0 = rc * 32;
  const int tid = threadIdx.x, wave = tid >> 5, lane = tid & 31;
  const int wr  = wave & 1;          // 16-row tile
  const int wcq = wave >> 1;         // 0..3
  const int m_lane = lane & 15;
  const int kb = (lane >> 4) << 3;
  const float lam = lam_buf[layer];
  const float scale = 0.17677669529663687f;   // 32^-0.5
  const size_t base_row = (size_t)b * TOK;
  const int c1 = (2 * h) * HDIM;
  const int c2 = c1 + HDIM;
  const int PSTR = NSEQP + 8;

  // Q fragments for this wave's row tile (both query maps)
  Frag aq1, aq2;
  {
    int qr = row0 + wr * 16 + m_lane;
    if (qr < TOK) {
      const unsigned short* p1 = Q + (base_row + qr) * DMODEL + c1 + kb;
      aq1.q[0] = *(const u32x4*)p1;
      aq1.q[1] = *(const u32x4*)(p1 + 16);
      const unsigned short* p2 = Q + (base_row + qr) * DMODEL + c2 + kb;
      aq2.q[0] = *(const u32x4*)p2;
      aq2.q[1] = *(const u32x4*)(p2 + 16);
    } else {
      aq1.q[0] = (u32x4)0u; aq1.q[1] = (u32x4)0u;
      aq2.q[0] = (u32x4)0u; aq2.q[1] = (u32x4)0u;
    }
  }

  for (int which = 0; which < 2; ++which) {
    const Frag& aq = which ? aq2 : aq1;
    const int ch = which ? c2 : c1;
    // scores: K dim = HDIM = 32 -> single WMMA per 16x16 tile
    for (int nt = wcq; nt < NSEQP / 16; nt += 4) {
      Frag bk;
      int kr = nt * 16 + m_lane;
      if (kr < TOK) {
        const unsigned short* p = K + (base_row + kr) * DMODEL + ch + kb;
        bk.q[0] = *(const u32x4*)p;
        bk.q[1] = *(const u32x4*)(p + 16);
      } else {
        bk.q[0] = (u32x4)0u; bk.q[1] = (u32x4)0u;
      }
      v8f acc = (v8f)0.0f;
      acc = wmma_bf16(aq, bk, acc);
      int ncol = nt * 16 + m_lane;
      int mrow = wr * 16 + ((lane >> 4) << 3);
#pragma unroll
      for (int r = 0; r < 8; ++r)
        sS[(mrow + r) * NSEQP + ncol] = acc[r] * scale;
    }
    __syncthreads();
    // softmax per row (8 lanes per row), then combine p1 - lam*p2 into sP (bf16)
    {
      int trow = tid >> 3, tg = tid & 7;
      float* rowp = sS + trow * NSEQP;
      float mx = -1e30f;
      for (int j = tg; j < TOK; j += 8) mx = fmaxf(mx, rowp[j]);
#pragma unroll
      for (int off = 4; off; off >>= 1) mx = fmaxf(mx, __shfl_xor(mx, off, 8));
      float sum = 0.0f;
      for (int j = tg; j < NSEQP; j += 8) {
        float e = (j < TOK) ? __expf(rowp[j] - mx) : 0.0f;
        rowp[j] = e;
        sum += e;
      }
#pragma unroll
      for (int off = 4; off; off >>= 1) sum += __shfl_xor(sum, off, 8);
      float inv = 1.0f / sum;
      unsigned short* pp = sP + trow * PSTR;
      if (which == 0) {
        for (int j = tg; j < NSEQP; j += 8) pp[j] = f32_to_bf16(rowp[j] * inv);
      } else {
        for (int j = tg; j < NSEQP; j += 8) {
          float p1 = bf16_to_f32(pp[j]);
          pp[j] = f32_to_bf16(p1 - lam * (rowp[j] * inv));
        }
      }
    }
    __syncthreads();
  }

  // O = attn[32,224] @ V[224,64] : wave -> one 16x16 tile, 7 WMMA K-steps
  v8f oacc = (v8f)0.0f;
  for (int ks = 0; ks < NSEQP / 32; ++ks) {
    Frag ap;
    const unsigned short* pp = sP + (wr * 16 + m_lane) * PSTR + ks * 32 + kb;
    ap.q[0] = *(const u32x4*)pp;
    ap.q[1] = *(const u32x4*)(pp + 16);
    Frag bv;
    int nc = wcq * 16 + m_lane;
#pragma unroll
    for (int i = 0; i < 16; ++i) {
      int kk = ks * 32 + kb + (i & 7) + ((i >> 3) << 4);
      bv.u[i] = (kk < TOK) ? V[(base_row + kk) * DMODEL + h * 64 + nc]
                           : (unsigned short)0;
    }
    oacc = wmma_bf16(ap, bv, oacc);
  }
  {
    int mrow = wr * 16 + ((lane >> 4) << 3);
#pragma unroll
    for (int r = 0; r < 8; ++r)
      sO[(mrow + r) * 64 + wcq * 16 + m_lane] = oacc[r];
  }
  __syncthreads();
  // RMSNorm over 64 dims, * subln, * (1 - lam_init); write bf16
  {
    int trow = tid >> 3, tg = tid & 7;
    const float* orow = sO + trow * 64;
    float ss = 0.0f;
    for (int j = tg; j < 64; j += 8) { float v = orow[j]; ss += v * v; }
#pragma unroll
    for (int off = 4; off; off >>= 1) ss += __shfl_xor(ss, off, 8);
    float inv = rsqrtf(ss * (1.0f / 64.0f) + 1e-5f) * (1.0f - lam_init);
    int grow = row0 + trow;
    if (grow < TOK) {
      unsigned short* op = Out + (base_row + grow) * DMODEL + h * 64;
      for (int j = tg; j < 64; j += 8)
        op[j] = f32_to_bf16(orow[j] * inv * subln[j]);
    }
  }
}

// ---------------- host side ----------------
static void launch_gemm(const unsigned short* A, const unsigned short* B,
                        const float* bias, const float* res,
                        float* Cf, unsigned short* Cbf,
                        int M, int N, int Kd, int act, hipStream_t s) {
  dim3 grid((N + GT_BN - 1) / GT_BN, (M + GT_BM - 1) / GT_BM);
  gemm_bf16_kernel<<<grid, 256, 0, s>>>(A, B, bias, res, Cf, Cbf, M, N, Kd, act);
}

extern "C" void kernel_launch(void* const* d_in, const int* in_sizes, int n_in,
                              void* d_out, int out_size, void* d_ws, size_t ws_size,
                              hipStream_t stream) {
  (void)in_sizes; (void)n_in; (void)out_size; (void)ws_size;

  const float* x      = (const float*)d_in[0];
  const float* conv_w = (const float*)d_in[1];
  const float* conv_b = (const float*)d_in[2];
  const float* cls    = (const float*)d_in[3];
  const float* pos    = (const float*)d_in[4];
  const float* ln1_s  = (const float*)d_in[5];
  const float* ln1_b  = (const float*)d_in[6];
  const float* q_w    = (const float*)d_in[7];
  const float* k_w    = (const float*)d_in[8];
  const float* v_w    = (const float*)d_in[9];
  const float* o_w    = (const float*)d_in[10];
  const float* lq1    = (const float*)d_in[11];
  const float* lk1    = (const float*)d_in[12];
  const float* lq2    = (const float*)d_in[13];
  const float* lk2    = (const float*)d_in[14];
  const float* subln  = (const float*)d_in[15];
  const float* ln2_s  = (const float*)d_in[16];
  const float* ln2_b  = (const float*)d_in[17];
  const float* w1     = (const float*)d_in[18];
  const float* b1     = (const float*)d_in[19];
  const float* w2     = (const float*)d_in[20];
  const float* b2     = (const float*)d_in[21];
  const float* lnf_s  = (const float*)d_in[22];
  const float* lnf_b  = (const float*)d_in[23];
  const float* head_w = (const float*)d_in[24];
  const float* head_b = (const float*)d_in[25];

  char* ws = (char*)d_ws;
  size_t off = 0;
  auto alloc = [&](size_t bytes) {
    size_t o = off;
    off = (o + bytes + 255) & ~(size_t)255;
    return o;
  };

  float*          h_buf  = (float*)         (ws + alloc((size_t)ROWS_TOTAL * DMODEL * 4));
  unsigned short* xn     = (unsigned short*)(ws + alloc((size_t)ROWS_TOTAL * DMODEL * 2));
  unsigned short* qb     = (unsigned short*)(ws + alloc((size_t)ROWS_TOTAL * DMODEL * 2));
  unsigned short* kbuf   = (unsigned short*)(ws + alloc((size_t)ROWS_TOTAL * DMODEL * 2));
  unsigned short* vbuf   = (unsigned short*)(ws + alloc((size_t)ROWS_TOTAL * DMODEL * 2));
  unsigned short* ao     = (unsigned short*)(ws + alloc((size_t)ROWS_TOTAL * DMODEL * 2));
  unsigned short* m1     = (unsigned short*)(ws + alloc((size_t)ROWS_TOTAL * MLPD * 2));
  unsigned short* wq_bf  = (unsigned short*)(ws + alloc((size_t)DMODEL * DMODEL * 2));
  unsigned short* wk_bf  = (unsigned short*)(ws + alloc((size_t)DMODEL * DMODEL * 2));
  unsigned short* wv_bf  = (unsigned short*)(ws + alloc((size_t)DMODEL * DMODEL * 2));
  unsigned short* wo_bf  = (unsigned short*)(ws + alloc((size_t)DMODEL * DMODEL * 2));
  unsigned short* w1_bf  = (unsigned short*)(ws + alloc((size_t)DMODEL * MLPD * 2));
  unsigned short* w2_bf  = (unsigned short*)(ws + alloc((size_t)DMODEL * MLPD * 2));
  unsigned short* convwT = (unsigned short*)(ws + alloc((size_t)DMODEL * DMODEL * 2));
  unsigned short* headwb = (unsigned short*)(ws + alloc((size_t)DMODEL * NCLS * 2));
  unsigned short* pein   = (unsigned short*)(ws + alloc((size_t)PATCH_ROWS * DMODEL * 2));
  float*          peout  = (float*)         (ws + alloc((size_t)PATCH_ROWS * DMODEL * 4));
  float*          lam    = (float*)         (ws + alloc((size_t)NLAYER * 4));
  float*          clsf   = (float*)         (ws + alloc((size_t)BATCH * DMODEL * 4));
  unsigned short* clsb   = (unsigned short*)(ws + alloc((size_t)BATCH * DMODEL * 2));

  // ---- setup: patch embed + token assembly + lambdas ----
  transpose_bf16_kernel<<<(DMODEL * DMODEL + 255) / 256, 256, 0, stream>>>(
      conv_w, convwT, DMODEL, DMODEL);
  convert_bf16_kernel<<<1024, 256, 0, stream>>>(head_w, headwb, (long long)DMODEL * NCLS);
  im2col_kernel<<<PATCH_ROWS, 256, 0, stream>>>(x, pein);
  launch_gemm(pein, convwT, conv_b, nullptr, peout, nullptr,
              PATCH_ROWS, DMODEL, DMODEL, 0, stream);
  {
    long long total = (long long)ROWS_TOTAL * DMODEL;
    assemble_h_kernel<<<(unsigned)((total + 255) / 256), 256, 0, stream>>>(
        peout, cls, pos, h_buf);
  }
  lambda_kernel<<<NLAYER, 32, 0, stream>>>(lq1, lk1, lq2, lk2, lam);

  // ---- transformer layers ----
  for (int i = 0; i < NLAYER; ++i) {
    size_t wdd = (size_t)i * DMODEL * DMODEL;
    size_t wdm = (size_t)i * DMODEL * MLPD;
    convert_bf16_kernel<<<1024, 256, 0, stream>>>(q_w + wdd, wq_bf, (long long)DMODEL * DMODEL);
    convert_bf16_kernel<<<1024, 256, 0, stream>>>(k_w + wdd, wk_bf, (long long)DMODEL * DMODEL);
    convert_bf16_kernel<<<1024, 256, 0, stream>>>(v_w + wdd, wv_bf, (long long)DMODEL * DMODEL);
    convert_bf16_kernel<<<1024, 256, 0, stream>>>(o_w + wdd, wo_bf, (long long)DMODEL * DMODEL);
    convert_bf16_kernel<<<2048, 256, 0, stream>>>(w1 + wdm, w1_bf, (long long)DMODEL * MLPD);
    convert_bf16_kernel<<<2048, 256, 0, stream>>>(w2 + wdm, w2_bf, (long long)DMODEL * MLPD);

    layernorm_kernel<<<ROWS_TOTAL, 256, 0, stream>>>(
        h_buf, ln1_s + (size_t)i * DMODEL, ln1_b + (size_t)i * DMODEL, xn);
    launch_gemm(xn, wq_bf, nullptr, nullptr, nullptr, qb,   ROWS_TOTAL, DMODEL, DMODEL, 0, stream);
    launch_gemm(xn, wk_bf, nullptr, nullptr, nullptr, kbuf, ROWS_TOTAL, DMODEL, DMODEL, 0, stream);
    launch_gemm(xn, wv_bf, nullptr, nullptr, nullptr, vbuf, ROWS_TOTAL, DMODEL, DMODEL, 0, stream);

    float lam_init = 0.8f - 0.6f * expf(-0.3f * (float)i);
    diff_attn_kernel<<<dim3((TOK + 31) / 32, NHEAD, BATCH), 256, 0, stream>>>(
        qb, kbuf, vbuf, lam, subln + (size_t)i * 2 * HDIM, lam_init, i, ao);

    launch_gemm(ao, wo_bf, nullptr, h_buf, h_buf, nullptr, ROWS_TOTAL, DMODEL, DMODEL, 0, stream);

    layernorm_kernel<<<ROWS_TOTAL, 256, 0, stream>>>(
        h_buf, ln2_s + (size_t)i * DMODEL, ln2_b + (size_t)i * DMODEL, xn);
    launch_gemm(xn, w1_bf, b1 + (size_t)i * MLPD, nullptr, nullptr, m1,
                ROWS_TOTAL, MLPD, DMODEL, 1, stream);
    launch_gemm(m1, w2_bf, b2 + (size_t)i * DMODEL, h_buf, h_buf, nullptr,
                ROWS_TOTAL, DMODEL, MLPD, 0, stream);
  }

  // ---- head ----
  gather_cls_kernel<<<(BATCH * DMODEL + 255) / 256, 256, 0, stream>>>(h_buf, clsf);
  layernorm_kernel<<<BATCH, 256, 0, stream>>>(clsf, lnf_s, lnf_b, clsb);
  launch_gemm(clsb, headwb, head_b, nullptr, (float*)d_out, nullptr,
              BATCH, NCLS, DMODEL, 0, stream);
}
